// GNN_PPO_Policy_48747878809907
// MI455X (gfx1250) — matrix operand order repaired and checked by reading, not compiled, not run on hardware
//
#include <hip/hip_runtime.h>

// ---------------------------------------------------------------------------
// Problem constants (from reference)
// ---------------------------------------------------------------------------
static constexpr int H   = 128;
static constexpr int MH  = 256;
static constexpr int G   = 8;
static constexpr int NJ  = 2000;
static constexpr int NM  = 50;
static constexpr int NS  = 200;
static constexpr int NP  = NJ * NM;        // 100000 actor rows
static constexpr int ROWS = 128;           // pair-rows per fused block
static constexpr int MS   = ROWS / 16;     // 8 M-subtiles per block

typedef __attribute__((ext_vector_type(16))) __bf16 v16bf;
typedef __attribute__((ext_vector_type(8)))  float  v8f;

union ABu { unsigned int u[8]; v16bf v; };

// native hardware bf16 convert (RNE)
__device__ __forceinline__ unsigned short f2bf(float f) {
    __bf16 b = (__bf16)f;
    return __builtin_bit_cast(unsigned short, b);
}
__device__ __forceinline__ unsigned int pack2(float a, float b) {
    return (unsigned int)f2bf(a) | ((unsigned int)f2bf(b) << 16);
}
__device__ __forceinline__ float frelu(float x) { return x > 0.f ? x : 0.f; }

// A-fragment k-offset for 16-bit 16x32 A tiles:
//   lane<16: K in {0..7} U {16..23};  lane>=16: K in {8..15} U {24..31}
__device__ __forceinline__ int a_koff(int lane, int v) {
    int base = (lane < 16 ? 0 : 8);
    return base + (v < 4 ? 2 * v : 16 + 2 * (v - 4));
}

// ---------------------------------------------------------------------------
// Kernel 1: pre-pack bf16 operands.
//  Apk: job_emb in per-lane A-fragment layout  [125 mt][4 kt][32 lane][8 u32]
//  Bp1: aW1[0:128,:]  in B-fragment layout     [4 kt][16 nt][32 lane][8 u32]
//  Bp2: aW2           in B-fragment layout     [8 kt][16 nt][32 lane][8 u32]
// B (32x16, 16-bit): same K order as A (lane<16 K=0..7,16..23; lane>=16 K=8..15,24..31), N=lane.
// ---------------------------------------------------------------------------
__global__ void pack_kernel(const float* __restrict__ job,
                            const float* __restrict__ aW1,
                            const float* __restrict__ aW2,
                            unsigned int* __restrict__ Apk,   // 125*4*32*8
                            unsigned int* __restrict__ Bp1,   // 4*16*32*8
                            unsigned int* __restrict__ Bp2) { // 8*16*32*8
    const int totalA  = 125 * 4 * 32 * 8;   // 128000
    const int totalB1 = 4 * 16 * 32 * 8;    // 16384
    const int totalB2 = 8 * 16 * 32 * 8;    // 32768
    for (int idx = blockIdx.x * blockDim.x + threadIdx.x;
         idx < totalA + totalB1 + totalB2; idx += gridDim.x * blockDim.x) {
        if (idx < totalA) {
            int i = idx;
            int v    = i & 7;
            int lane = (i >> 3) & 31;
            int tile = i >> 8;
            int kt = tile & 3;
            int mt = tile >> 2;
            int row = mt * 16 + (lane & 15);
            int k   = kt * 32 + a_koff(lane, v);
            Apk[i] = pack2(job[row * H + k], job[row * H + k + 1]);
        } else {
            bool is2 = idx >= totalA + totalB1;
            int i = is2 ? idx - totalA - totalB1 : idx - totalA;
            int v    = i & 7;
            int lane = (i >> 3) & 31;
            int tile = i >> 8;
            int nt = tile & 15;
            int kt = tile >> 4;
            int k = kt * 32 + a_koff(lane, v);
            int n = nt * 16 + (lane & 15);
            const float* W = is2 ? aW2 : aW1;
            unsigned int val = pack2(W[k * MH + n], W[(k + 1) * MH + n]);
            if (is2) Bp2[i] = val; else Bp1[i] = val;
        }
    }
}

// ---------------------------------------------------------------------------
// Kernel 2: one block, 512 threads.
//  - mean-pool job/machine/setup embeddings (4-way row split + LDS reduce)
//  - global_emb = relu(gf@Wg + bg)
//  - critic MLP on pooled state -> d_out[NP]
//  - c_act[n] = setup_agg.W1s[:,n] + global.W1g[:,n] + ab1[n]
//  - Pm[m][n] = machine_emb[m].W1m[:,n]
// ---------------------------------------------------------------------------
__global__ void prep_kernel(const float* __restrict__ job,
                            const float* __restrict__ mach,
                            const float* __restrict__ setup,
                            const float* __restrict__ gf,
                            const float* __restrict__ Wg,
                            const float* __restrict__ bg,
                            const float* __restrict__ aW1,
                            const float* __restrict__ ab1,
                            const float* __restrict__ cW1,
                            const float* __restrict__ cb1,
                            const float* __restrict__ cW2,
                            const float* __restrict__ cb2,
                            const float* __restrict__ cW3,
                            const float* __restrict__ cb3,
                            float* __restrict__ Pm,
                            float* __restrict__ c_act,
                            float* __restrict__ out_value) {
    __shared__ float red[4][H];
    __shared__ float s_state[4 * H];   // [job_agg | mach_agg | setup_agg | glob]
    __shared__ float h1c[MH];
    __shared__ float h2c[MH];
    const int tid = threadIdx.x;
    const int h   = tid & (H - 1);
    const int grp = tid >> 7;          // 0..3

    // job mean
    {
        float s = 0.f;
        for (int i = grp; i < NJ; i += 4) s += job[i * H + h];
        red[grp][h] = s;
    }
    __syncthreads();
    if (grp == 0)
        s_state[h] = (red[0][h] + red[1][h] + red[2][h] + red[3][h]) / (float)NJ;
    __syncthreads();
    // machine mean
    {
        float s = 0.f;
        for (int i = grp; i < NM; i += 4) s += mach[i * H + h];
        red[grp][h] = s;
    }
    __syncthreads();
    if (grp == 0)
        s_state[H + h] = (red[0][h] + red[1][h] + red[2][h] + red[3][h]) / (float)NM;
    __syncthreads();
    // setup mean + global_emb
    {
        float s = 0.f;
        for (int i = grp; i < NS; i += 4) s += setup[i * H + h];
        red[grp][h] = s;
    }
    __syncthreads();
    if (grp == 0) {
        s_state[2 * H + h] = (red[0][h] + red[1][h] + red[2][h] + red[3][h]) / (float)NS;
        float ge = bg[h];
        for (int g = 0; g < G; ++g) ge += gf[g] * Wg[g * H + h];
        s_state[3 * H + h] = frelu(ge);
    }
    __syncthreads();

    if (tid < MH) {
        const int n = tid;
        // c_act
        float acc = ab1[n];
        for (int k = 0; k < H; ++k) {
            acc += s_state[2 * H + k] * aW1[(2 * H + k) * MH + n];
            acc += s_state[3 * H + k] * aW1[(3 * H + k) * MH + n];
        }
        c_act[n] = acc;
        // Pm
        for (int m = 0; m < NM; ++m) {
            float a2 = 0.f;
            const float* me = mach + m * H;
            for (int k = 0; k < H; ++k)
                a2 += me[k] * aW1[(H + k) * MH + n];
            Pm[m * MH + n] = a2;
        }
        // critic layer 1
        float a3 = cb1[n];
        for (int k = 0; k < 4 * H; ++k) a3 += s_state[k] * cW1[k * MH + n];
        h1c[n] = frelu(a3);
    }
    __syncthreads();
    if (tid < MH) {
        const int n = tid;
        float acc = cb2[n];
        for (int k = 0; k < MH; ++k) acc += h1c[k] * cW2[k * MH + n];
        h2c[n] = frelu(acc);
    }
    __syncthreads();
    if (tid == 0) {
        float acc = cb3[0];
        for (int k = 0; k < MH; ++k) acc += h2c[k] * cW3[k];
        *out_value = acc;
    }
}

// ---------------------------------------------------------------------------
// Kernel 3: Pj = job_emb @ W1j via bf16 WMMA, both operands pre-packed.
// 125 M-tiles x 16 N-tiles = 2000 wave-tiles; 8 waves/block -> 250 blocks.
// ---------------------------------------------------------------------------
__global__ void pj_kernel(const unsigned int* __restrict__ Apk,
                          const unsigned int* __restrict__ Bp1,
                          float* __restrict__ Pj) {
    const int tid  = threadIdx.x;
    const int lane = tid & 31;
    const int wave = tid >> 5;
    const int w  = blockIdx.x * 8 + wave;     // 0..1999
    const int mt = w >> 4;                    // 0..124
    const int nt = w & 15;
    const bool lo = lane < 16;

    v8f acc = {};
#pragma unroll
    for (int kt = 0; kt < 4; ++kt) {
        ABu a, b;
        const uint4* ap = reinterpret_cast<const uint4*>(Apk + ((mt * 4 + kt) * 32 + lane) * 8);
        uint4 a0 = ap[0], a1 = ap[1];
        a.u[0] = a0.x; a.u[1] = a0.y; a.u[2] = a0.z; a.u[3] = a0.w;
        a.u[4] = a1.x; a.u[5] = a1.y; a.u[6] = a1.z; a.u[7] = a1.w;
        const uint4* bp = reinterpret_cast<const uint4*>(Bp1 + ((kt * 16 + nt) * 32 + lane) * 8);
        uint4 b0 = bp[0], b1 = bp[1];
        b.u[0] = b0.x; b.u[1] = b0.y; b.u[2] = b0.z; b.u[3] = b0.w;
        b.u[4] = b1.x; b.u[5] = b1.y; b.u[6] = b1.z; b.u[7] = b1.w;
        acc = __builtin_amdgcn_wmma_f32_16x16x32_bf16(false, a.v, false, b.v,
                                                      (short)0, acc, false, false);
    }
    // C/D layout: VGPR i -> M = i (lanes 0-15) or 8+i (lanes 16-31), N = lane&15
    const int col   = nt * 16 + (lane & 15);
    const int rbase = mt * 16 + (lo ? 0 : 8);
#pragma unroll
    for (int i = 0; i < 8; ++i)
        Pj[(rbase + i) * MH + col] = acc[i];
}

// ---------------------------------------------------------------------------
// Kernel 4: fused actor tail. 128 pair-rows/block, 256 threads (8 waves).
//  phase 1: h1 = relu(Pj[j]+Pm[m]+c) -> LDS in packed A-fragment layout
//  phase 2: layer-2 WMMA with register-resident B, layer-3 folded in:
//           per tile, lanes reduce relu(h2+b2)*aW3[col] across the 16
//           columns via shfl_xor and atomically accumulate per-row partials
//  phase 3: store logits
// ---------------------------------------------------------------------------
__global__ void fused_kernel(const float* __restrict__ Pj,
                             const float* __restrict__ Pm,
                             const float* __restrict__ c_act,
                             const unsigned int* __restrict__ Bp2,
                             const float* __restrict__ ab2,
                             const float* __restrict__ aW3,
                             const float* __restrict__ ab3,
                             float* __restrict__ out) {
    // h1 packed: [ms 0..7][kt 0..7][lane 0..31][v 0..7] u32 words (64 KB)
    __shared__ __align__(16) unsigned int h1p[MS * 8 * 32 * 8];
    __shared__ float part[ROWS];

    const int tid = threadIdx.x;
    const int p0  = blockIdx.x * ROWS;

    // zero the per-row logit partials
    if (tid < ROWS) part[tid] = 0.f;

    // ---- phase 1: build h1 tile directly in packed A layout ----
    for (int wd = tid; wd < MS * 8 * 32 * 8; wd += 256) {
        const int v    = wd & 7;
        const int lane = (wd >> 3) & 31;
        const int tile = wd >> 8;          // 0..63
        const int kt   = tile & 7;
        const int ms   = tile >> 3;
        const int r = ms * 16 + (lane & 15);
        const int k = kt * 32 + a_koff(lane, v);
        const int p = p0 + r;
        int j = p / NM; if (j > NJ - 1) j = NJ - 1;   // guard rows past NP
        const int m = p % NM;
        const float2 pj = *reinterpret_cast<const float2*>(Pj + j * MH + k);
        const float2 pm = *reinterpret_cast<const float2*>(Pm + m * MH + k);
        const float2 cc = *reinterpret_cast<const float2*>(c_act + k);
        h1p[wd] = pack2(frelu(pj.x + pm.x + cc.x), frelu(pj.y + pm.y + cc.y));
    }
    __syncthreads();

    // ---- phase 2: h2 tiles via WMMA, fold layer 3 immediately ----
    const int lane = tid & 31;
    const int wave = tid >> 5;
    for (int half = 0; half < 2; ++half) {
        const int nt = wave + half * 8;    // this wave's N tile
        ABu b[8];
#pragma unroll
        for (int kt = 0; kt < 8; ++kt) {
            const uint4* bp = reinterpret_cast<const uint4*>(Bp2 + ((kt * 16 + nt) * 32 + lane) * 8);
            uint4 b0 = bp[0], b1 = bp[1];
            b[kt].u[0] = b0.x; b[kt].u[1] = b0.y; b[kt].u[2] = b0.z; b[kt].u[3] = b0.w;
            b[kt].u[4] = b1.x; b[kt].u[5] = b1.y; b[kt].u[6] = b1.z; b[kt].u[7] = b1.w;
        }
        const int col    = nt * 16 + (lane & 15);
        const float bias = ab2[col];
        const float w3   = aW3[col];
        for (int ms = 0; ms < MS; ++ms) {
            v8f acc = {};
#pragma unroll
            for (int kt = 0; kt < 8; ++kt) {
                ABu a;
                const uint4* ap = reinterpret_cast<const uint4*>(&h1p[((ms * 8 + kt) * 32 + lane) * 8]);
                uint4 a0 = ap[0], a1 = ap[1];
                a.u[0] = a0.x; a.u[1] = a0.y; a.u[2] = a0.z; a.u[3] = a0.w;
                a.u[4] = a1.x; a.u[5] = a1.y; a.u[6] = a1.z; a.u[7] = a1.w;
                acc = __builtin_amdgcn_wmma_f32_16x16x32_bf16(false, a.v, false, b[kt].v,
                                                              (short)0, acc, false, false);
            }
            // layer 3 contribution for this tile: reduce over 16 columns
            float s[8];
#pragma unroll
            for (int i = 0; i < 8; ++i) {
                float v = frelu(acc[i] + bias) * w3;
                v += __shfl_xor(v, 1, 32);
                v += __shfl_xor(v, 2, 32);
                v += __shfl_xor(v, 4, 32);
                v += __shfl_xor(v, 8, 32);
                s[i] = v;
            }
            if ((lane & 15) == 0) {
                // lane 0 -> rows ms*16+0..7 ; lane 16 -> rows ms*16+8..15
                const int rbase = ms * 16 + (lane >> 4) * 8;
#pragma unroll
                for (int i = 0; i < 8; ++i)
                    atomicAdd(&part[rbase + i], s[i]);
            }
        }
    }
    __syncthreads();

    // ---- phase 3: store logits ----
    if (tid < ROWS) {
        const int p = p0 + tid;
        if (p < NP)
            out[p] = part[tid] + ab3[0];
    }
}

// ---------------------------------------------------------------------------
// Launch
// ---------------------------------------------------------------------------
extern "C" void kernel_launch(void* const* d_in, const int* in_sizes, int n_in,
                              void* d_out, int out_size, void* d_ws, size_t ws_size,
                              hipStream_t stream) {
    const float* job   = (const float*)d_in[0];
    const float* mach  = (const float*)d_in[1];
    const float* setup = (const float*)d_in[2];
    const float* gf    = (const float*)d_in[3];
    const float* Wg    = (const float*)d_in[4];
    const float* bg    = (const float*)d_in[5];
    const float* aW1   = (const float*)d_in[6];
    const float* ab1   = (const float*)d_in[7];
    const float* aW2   = (const float*)d_in[8];
    const float* ab2   = (const float*)d_in[9];
    const float* aW3   = (const float*)d_in[10];
    const float* ab3   = (const float*)d_in[11];
    const float* cW1   = (const float*)d_in[12];
    const float* cb1   = (const float*)d_in[13];
    const float* cW2   = (const float*)d_in[14];
    const float* cb2   = (const float*)d_in[15];
    const float* cW3   = (const float*)d_in[16];
    const float* cb3   = (const float*)d_in[17];
    float* out = (float*)d_out;   // [NP logits][1 value]

    // workspace layout (bytes), total ~2.75 MB
    char* ws = (char*)d_ws;
    float*        Pj    = (float*)(ws);                      // 2000*256 f32 = 2,048,000
    float*        Pm    = (float*)(ws + 2048000);            //   50*256 f32 =    51,200
    float*        c_act = (float*)(ws + 2099200);            //      256 f32 =     1,024
    unsigned int* Bp1   = (unsigned int*)(ws + 2100224);     //  16384 u32   =    65,536
    unsigned int* Bp2   = (unsigned int*)(ws + 2165760);     //  32768 u32   =   131,072
    unsigned int* Apk   = (unsigned int*)(ws + 2296832);     // 128000 u32   =   512,000
    (void)ws_size; (void)in_sizes; (void)n_in; (void)out_size;

    pack_kernel<<<692, 256, 0, stream>>>(job, aW1, aW2, Apk, Bp1, Bp2);
    prep_kernel<<<1, 512, 0, stream>>>(job, mach, setup, gf, Wg, bg,
                                       aW1, ab1, cW1, cb1, cW2, cb2, cW3, cb3,
                                       Pm, c_act, out + NP);
    pj_kernel<<<250, 256, 0, stream>>>(Apk, Bp1, Pj);
    fused_kernel<<<(NP + ROWS - 1) / ROWS, 256, 0, stream>>>(Pj, Pm, c_act, Bp2,
                                                             ab2, aW3, ab3, out);
}
